// BasicDetectionModel_50551765074321
// MI455X (gfx1250) — matrix operand order repaired and checked by reading, not compiled, not run on hardware
//
#include <hip/hip_runtime.h>
#include <hip/hip_bf16.h>

// ---------------------------------------------------------------------------
// CDNA5 / gfx1250 BasicDetectionModel forward loss.
//  - convs: implicit GEMM on v_wmma_f32_16x16x32_bf16, inputs staged per-block
//    into LDS by the Tensor Data Mover (tensor_load_to_lds + s_wait_tensorcnt),
//    weights zero-padded bf16 in LDS, im2col offsets via LDS lookup table.
//  - heads + IoU assignment + CE + smooth-L1 fused (logits live in LDS only),
//    head weights packed/zero-padded in LDS so fragment builds are branch-free.
// ---------------------------------------------------------------------------

typedef __bf16 bf16_t;
typedef __attribute__((ext_vector_type(16))) __bf16 v16bf;
typedef __attribute__((ext_vector_type(8)))  float  v8f;
typedef unsigned int u32x4 __attribute__((ext_vector_type(4)));
typedef int          i32x4 __attribute__((ext_vector_type(4)));
typedef int          i32x8 __attribute__((ext_vector_type(8)));

#define NUM_CLASSES 80
#define GT_PER_IMG  32

#ifndef USE_TDM
#define USE_TDM 1
#endif

#if USE_TDM && defined(__HIP_DEVICE_COMPILE__)
#if !__has_builtin(__builtin_amdgcn_tensor_load_to_lds)
#undef USE_TDM
#define USE_TDM 0
#endif
#endif

// ---------------------------------------------------------------------------
// Conv 3x3 stride 2 pad 1 + bias + ReLU.  GEMM: C[COUT x 256px] = W[COUT x K]
// * im2col[K x 256px], K = CIN*9.  Block = 256 threads (8 waves), computes a
// 16x16 output tile for all COUT channels.  Per wave-step: one 16(M)x16(N)
// WMMA C tile, K swept in 32-wide bf16 steps.
// Fragments: A 16x32 / B 32x16 both with contiguous k = 16*hi + e; C 8xf32 per CDNA5 ISA 7.12.2.
// ---------------------------------------------------------------------------
template <int CIN, int COUT, typename InT>
__global__ __launch_bounds__(256) void conv3x3_s2_relu_wmma(
    const InT*   __restrict__ in,    // [B, CIN, Hin, Win]
    const float* __restrict__ w,     // [COUT, CIN, 3, 3] == [COUT, K]
    const float* __restrict__ bias,  // [COUT]
    bf16_t*      __restrict__ out,   // [B, COUT, Hout, Wout]
    int Hin, int Win, int Hout, int Wout)
{
  constexpr int K      = CIN * 9;
  constexpr int KSTEPS = (K + 31) / 32;
  constexpr int KP     = KSTEPS * 32;      // K padded to WMMA step
  constexpr int TH     = 33;               // input tile rows  (15*2+3)
  constexpr int TW     = 36;               // LDS row pitch (33 cols + pad)
  constexpr int PLANE  = TH * TW;          // per-channel LDS plane
  constexpr int MTILES = COUT / 16;

  __shared__ InT    s_in[CIN * PLANE + 40]; // +40: TDM edge-shift overflow slack
  __shared__ bf16_t s_wb[COUT * KP];        // weights, bf16, zero-padded K
  __shared__ int    s_off[KP];              // im2col k -> element offset in s_in

  const int tid  = threadIdx.x;
  const int lane = tid & 31;
  const int wave = tid >> 5;
  const int hi   = lane >> 4;
  const int l16  = lane & 15;
  const int b    = blockIdx.z;
  const int oh0  = blockIdx.y * 16;
  const int ow0  = blockIdx.x * 16;

  const long in_img = (long)b * CIN * Hin * Win;

  // top/left halo clamp: tile logically starts at (oh0*2-1, ow0*2-1)
  const int r0   = (oh0 == 0) ? 1 : 0;     // LDS row shift at top edge
  const int c0   = (ow0 == 0) ? 1 : 0;     // LDS col shift at left edge
  const int gih0 = oh0 * 2 - 1 + r0;       // first global row actually read
  const int giw0 = ow0 * 2 - 1 + c0;

#if USE_TDM
  // ---- async tile fetch: 3D TDM tile (x=36 cols, y=33 rows, z=CIN) ----
  if (wave == 0) {
    constexpr unsigned DSZ_CODE = (sizeof(InT) == 4) ? 2u : 1u;  // 4B / 2B
    const unsigned long long ga =
        (unsigned long long)(size_t)(in + in_img + (long)gih0 * Win + giw0);
    const unsigned lds =
        (unsigned)(size_t)(&s_in[0]) + (unsigned)((r0 * TW + c0) * sizeof(InT));
    const unsigned td0 = (unsigned)(Win - giw0);        // remaining extent x
    const unsigned td1 = (unsigned)(Hin - gih0);        // remaining extent y
    const unsigned long long st1 = (unsigned long long)Hin * (unsigned)Win;

    u32x4 g0 = {};
    g0[0] = 1u;                                          // count=1 user D#
    g0[1] = lds;                                         // lds_addr
    g0[2] = (unsigned)(ga & 0xFFFFFFFFull);              // global_addr[31:0]
    g0[3] = (unsigned)((ga >> 32) & 0x01FFFFFFull) | (2u << 30);  // addr + type=2

    i32x8 g1 = {};
    g1[0] = (int)(DSZ_CODE << 16);                       // data_size
    g1[1] = (int)((td0 & 0xFFFFu) << 16);                // tensor_dim0[15:0]
    g1[2] = (int)((td0 >> 16) | ((td1 & 0xFFFFu) << 16));// dim0 hi | dim1 lo
    g1[3] = (int)((td1 >> 16) | ((unsigned)TW << 16));   // dim1 hi | tile_dim0=36
    g1[4] = (int)((unsigned)TH | ((unsigned)CIN << 16)); // tile_dim1=33, tile_dim2
    g1[5] = (int)(unsigned)Win;                          // dim0_stride lo32
    g1[6] = (int)((unsigned)(st1 & 0xFFFFull) << 16);    // dim0_stride hi=0|dim1_stride lo16
    g1[7] = (int)(st1 >> 16);                            // dim1_stride[47:16]

    i32x4 g2 = {};
    g2[0] = CIN;                                         // tensor_dim2
    i32x4 g3 = {};
    i32x8 g4 = {};                                       // trailing group (zero)

    __builtin_amdgcn_tensor_load_to_lds(g0, g1, g2, g3, g4, 0);
  }
#else
  // ---- fallback: cooperative bounds-checked tile fill ----
  for (int idx = tid; idx < CIN * TH * TW; idx += 256) {
    const int ci = idx / PLANE;
    const int rr = idx - ci * PLANE;
    const int r  = rr / TW;
    const int c  = rr - r * TW;
    const int ih = oh0 * 2 - 1 + r;
    const int iw = ow0 * 2 - 1 + c;
    InT v = (InT)0.f;
    if ((unsigned)ih < (unsigned)Hin && (unsigned)iw < (unsigned)Win)
      v = in[in_img + ((long)ci * Hin + ih) * Win + iw];
    s_in[idx] = v;
  }
#endif

  // ---- weights -> LDS bf16, K zero-padded to KP (runs while TDM streams) ----
  for (int i = tid; i < COUT * KP; i += 256) {
    const int m = i / KP, k = i - m * KP;
    s_wb[i] = (bf16_t)((k < K) ? w[m * K + k] : 0.f);
  }
  // ---- im2col k -> LDS offset table ----
  for (int k = tid; k < KP; k += 256) {
    int off = 0;
    if (k < K) {
      const int ci = k / 9, r = k - ci * 9;
      off = ci * PLANE + (r / 3) * TW + (r % 3);
    }
    s_off[k] = off;
  }

#if USE_TDM
  if (wave == 0) __builtin_amdgcn_s_wait_tensorcnt(0);
  __syncthreads();
  // zero the halo rows/cols the clamped TDM did not cover (also fixes the
  // one-element per-plane overflow the shifted tile writes there)
  if (r0) for (int i = tid; i < CIN * TW; i += 256)
    s_in[(i / TW) * PLANE + (i % TW)] = (InT)0.f;
  if (c0) for (int i = tid; i < CIN * TH; i += 256)
    s_in[(i / TH) * PLANE + (i % TH) * TW] = (InT)0.f;
#endif
  __syncthreads();

  // ---- compute: MTILES*16 wave-tiles (M-tile x output-row), 8 waves ----
  for (int t = wave; t < MTILES * 16; t += 8) {
    const int mt   = t >> 4;                 // out-channel tile
    const int oh_l = t & 15;                 // output row within block tile
    const int m_a  = mt * 16 + l16;          // A row for this lane
    const int pixb = (oh_l * 2) * TW + l16 * 2;  // lane's window base in s_in

    v8f acc = {};
#pragma unroll
    for (int ks = 0; ks < KSTEPS; ++ks) {
      const int k0 = ks * 32;
      v16bf afrag, bfrag;
#pragma unroll
      for (int e = 0; e < 16; ++e) {
        const int kb = k0 + e + 16 * hi;                       // B element k
        const bf16_t vb = (bf16_t)s_in[pixb + s_off[kb]];
        bfrag[e] = (kb < K) ? vb : (bf16_t)0.f;
        const int ka = k0 + e + 16 * hi;                       // A element k
        afrag[e] = s_wb[m_a * KP + ka];                        // zero-padded
      }
      acc = __builtin_amdgcn_wmma_f32_16x16x32_bf16(
          false, afrag, false, bfrag, (short)0, acc, false, false);
    }

    const int oh = oh0 + oh_l;
#pragma unroll
    for (int i = 0; i < 8; ++i) {
      const int m = mt * 16 + i + 8 * hi;
      float v = acc[i] + bias[m];
      v = v > 0.f ? v : 0.f;
      out[(((long)b * COUT + m) * Hout + oh) * Wout + ow0 + l16] = (bf16_t)v;
    }
  }
}

// ---------------------------------------------------------------------------
// Fused heads + loss: cls 64->80 and reg 64->4 packed into a 96x64 bf16 weight
// block in LDS (zero-padded), 6 WMMA M-tiles per wave, logits staged in LDS,
// per-pixel IoU argmax + log-softmax CE + smooth-L1, LDS atomics + three global
// atomicAdds per block.
// ---------------------------------------------------------------------------
__global__ __launch_bounds__(256) void head_loss_wmma(
    const bf16_t* __restrict__ feat,  // [B, 64, P]
    const float*  __restrict__ wc, const float* __restrict__ bc,
    const float*  __restrict__ wr, const float* __restrict__ br,
    const float*  __restrict__ gtb,   // [B, 32, 4]
    const int*    __restrict__ gtl,   // [B, 32]
    float*        __restrict__ accum, // 0=cls_sum, 1..16=sl1[b], 17..32=cnt[b]
    int Hf, int Wf)
{
  const int P    = Hf * Wf;
  const int lane = threadIdx.x & 31;
  const int wave = threadIdx.x >> 5;
  const int hi   = lane >> 4;
  const int l16  = lane & 15;
  const int b    = blockIdx.z;
  const int blk_n0 = blockIdx.x * 128;
  const int n0     = blk_n0 + wave * 16;

  __shared__ float  logits[128][85];   // 84 rows used (80 cls + 4 reg)
  __shared__ bf16_t s_hw[96 * 64];     // packed head weights, zero-padded rows
  __shared__ float  s_hb[96];          // packed head bias
  __shared__ float  s_cls, s_sl1;
  __shared__ int    s_cnt;

  // ---- stage packed head weights/bias into LDS (branch-free fragments) ----
  for (int i = threadIdx.x; i < 96 * 64; i += 256) {
    const int m = i >> 6, k = i & 63;
    float v = 0.f;
    if (m < NUM_CLASSES)          v = wc[m * 64 + k];
    else if (m < NUM_CLASSES + 4) v = wr[(m - NUM_CLASSES) * 64 + k];
    s_hw[i] = (bf16_t)v;
  }
  for (int i = threadIdx.x; i < 96; i += 256) {
    s_hb[i] = (i < NUM_CLASSES) ? bc[i]
            : (i < NUM_CLASSES + 4) ? br[i - NUM_CLASSES] : 0.f;
  }
  if (threadIdx.x == 0) { s_cls = 0.f; s_sl1 = 0.f; s_cnt = 0; }
  __syncthreads();

  // ---- head GEMM: [96 x 64] x [64 x 16] per wave, 6 M-tiles x 2 K-steps ----
  v8f c[6];
  {
    const v8f z = {};
#pragma unroll
    for (int t = 0; t < 6; ++t) c[t] = z;
  }

  const long fbase = (long)b * 64 * P;
#pragma unroll
  for (int ks = 0; ks < 2; ++ks) {
    const int k0 = ks << 5;
    v16bf bfrag;
#pragma unroll
    for (int e = 0; e < 16; ++e) {
      const int k = k0 + e + 16 * hi;
      bfrag[e] = feat[fbase + (long)k * P + n0 + l16];
    }
#pragma unroll
    for (int t = 0; t < 6; ++t) {
      const int m = t * 16 + l16;
      v16bf afrag;
#pragma unroll
      for (int e = 0; e < 16; ++e) {
        const int k = k0 + e + 16 * hi;
        afrag[e] = s_hw[(m << 6) + k];
      }
      c[t] = __builtin_amdgcn_wmma_f32_16x16x32_bf16(
          false, afrag, false, bfrag, (short)0, c[t], false, false);
    }
  }

  // ---- spill logits (+bias) to LDS ----
#pragma unroll
  for (int t = 0; t < 6; ++t) {
#pragma unroll
    for (int i = 0; i < 8; ++i) {
      const int m = t * 16 + i + 8 * hi;
      if (m < NUM_CLASSES + 4)
        logits[wave * 16 + l16][m] = c[t][i] + s_hb[m];
    }
  }
  __syncthreads();

  // ---- per-pixel loss (128 pixels per block) ----
  if (threadIdx.x < 128) {
    const int q = threadIdx.x;
    const int n = blk_n0 + q;
    const float cx0 = (float)(n % Wf);
    const float cy0 = (float)(n / Wf);
    const float cx1 = cx0 + 1.f, cy1 = cy0 + 1.f;
    const float ac  = (cx1 - cx0) * (cy1 - cy0);

    const float* gb = gtb + (long)b * GT_PER_IMG * 4;
    float best = -1.f;
    int   bidx = 0;
    for (int g = 0; g < GT_PER_IMG; ++g) {
      const float gx0 = gb[g * 4 + 0], gy0 = gb[g * 4 + 1];
      const float gx1 = gb[g * 4 + 2], gy1 = gb[g * 4 + 3];
      const float ix0 = fmaxf(gx0, cx0), iy0 = fmaxf(gy0, cy0);
      const float ix1 = fminf(gx1, cx1), iy1 = fminf(gy1, cy1);
      const float iw = fmaxf(ix1 - ix0, 0.f), ih = fmaxf(iy1 - iy0, 0.f);
      const float inter = iw * ih;
      const float ag  = (gx1 - gx0) * (gy1 - gy0);
      const float iou = inter / (ag + ac - inter);
      if (iou > best) { best = iou; bidx = g; }   // first-max == jnp.argmax
    }
    const int assigned = (best < 0.5f) ? 0 : gtl[b * GT_PER_IMG + bidx];

    float mx = -1e30f;
    for (int m = 0; m < NUM_CLASSES; ++m) mx = fmaxf(mx, logits[q][m]);
    float se = 0.f;
    for (int m = 0; m < NUM_CLASSES; ++m) se += __expf(logits[q][m] - mx);
    const float ce = (__logf(se) + mx) - logits[q][assigned];
    atomicAdd(&s_cls, ce);

    if (assigned > 0) {
      float sl = 0.f;
#pragma unroll
      for (int j = 0; j < 4; ++j) {
        const float d = fabsf(logits[q][NUM_CLASSES + j] - gb[bidx * 4 + j]);
        sl += (d < 1.f) ? 0.5f * d * d : d - 0.5f;
      }
      atomicAdd(&s_sl1, sl);
      atomicAdd(&s_cnt, 1);
    }
  }
  __syncthreads();

  if (threadIdx.x == 0) {
    atomicAdd(&accum[0], s_cls);
    atomicAdd(&accum[1 + b], s_sl1);
    atomicAdd(&accum[17 + b], (float)s_cnt);
  }
}

// ---------------------------------------------------------------------------
__global__ void zero_accum_kernel(float* __restrict__ a)
{
  if (threadIdx.x < 64) a[threadIdx.x] = 0.f;
}

__global__ void finalize_losses(const float* __restrict__ accum,
                                float* __restrict__ out, int Bn, long P)
{
  if (threadIdx.x == 0 && blockIdx.x == 0) {
    out[0] = accum[0] / (float)((long)Bn * P);
    float s = 0.f;
    for (int b = 0; b < Bn; ++b) {
      const float cnt = accum[17 + b];
      s += (cnt > 0.f) ? accum[1 + b] / (4.f * fmaxf(cnt, 1.f)) : 0.f;
    }
    out[1] = s / (float)Bn;
  }
}

// ---------------------------------------------------------------------------
extern "C" void kernel_launch(void* const* d_in, const int* in_sizes, int n_in,
                              void* d_out, int out_size, void* d_ws, size_t ws_size,
                              hipStream_t stream)
{
  (void)in_sizes; (void)n_in; (void)out_size; (void)ws_size;

  const float* images    = (const float*)d_in[0];   // [16,3,640,640]
  const float* gt_boxes  = (const float*)d_in[1];   // [16,32,4]
  const int*   gt_labels = (const int*)  d_in[2];   // [16,32]
  const float* w1 = (const float*)d_in[3];
  const float* b1 = (const float*)d_in[4];
  const float* w2 = (const float*)d_in[5];
  const float* b2 = (const float*)d_in[6];
  const float* w3 = (const float*)d_in[7];
  const float* b3 = (const float*)d_in[8];
  const float* wc = (const float*)d_in[9];
  const float* bc = (const float*)d_in[10];
  const float* wr = (const float*)d_in[11];
  const float* br = (const float*)d_in[12];
  float* out = (float*)d_out;

  const int B = 16;

  char*   ws    = (char*)d_ws;
  float*  accum = (float*)ws;
  bf16_t* act1  = (bf16_t*)(ws + 256);               // [16,16,320,320] 52.4 MB
  bf16_t* act2  = act1 + (size_t)B * 16 * 320 * 320; // [16,32,160,160] 26.2 MB
  bf16_t* act3  = act2 + (size_t)B * 32 * 160 * 160; // [16,64, 80, 80] 13.1 MB

  zero_accum_kernel<<<1, 64, 0, stream>>>(accum);

  conv3x3_s2_relu_wmma<3, 16, float><<<dim3(20, 20, B), 256, 0, stream>>>(
      images, w1, b1, act1, 640, 640, 320, 320);

  conv3x3_s2_relu_wmma<16, 32, bf16_t><<<dim3(10, 10, B), 256, 0, stream>>>(
      act1, w2, b2, act2, 320, 320, 160, 160);

  conv3x3_s2_relu_wmma<32, 64, bf16_t><<<dim3(5, 5, B), 256, 0, stream>>>(
      act2, w3, b3, act3, 160, 160, 80, 80);

  head_loss_wmma<<<dim3(50, 1, B), 256, 0, stream>>>(
      act3, wc, bc, wr, br, gt_boxes, gt_labels, accum, 80, 80);

  finalize_losses<<<1, 32, 0, stream>>>(accum, out, B, 6400L);
}
